// CausalSelfAttention_7267084665085
// MI455X (gfx1250) — matrix operand order repaired and checked by reading, not compiled, not run on hardware
//
#include <hip/hip_runtime.h>

typedef __attribute__((ext_vector_type(16))) _Float16 v16h;
typedef __attribute__((ext_vector_type(8)))  float    v8f;

#define WMMA(A, B, C) \
  __builtin_amdgcn_wmma_f32_16x16x32_f16(false, (A), false, (B), (short)0, (C), false, false)

static constexpr int kD  = 1024;   // d_model
static constexpr int kH  = 16;     // heads
static constexpr int kDh = 64;     // head dim
static constexpr int kT  = 2048;   // seq len
static constexpr int kB  = 2;      // batch
static constexpr int kM  = kB * kT;        // 4096 token rows
static constexpr int k3D = 3 * kD;         // 3072

// ---------------------------------------------------------------------------
// CDNA5 async global->LDS copy (ASYNCcnt path, ISA §10 / §15.18.3).
// GVS addressing: mem = SADDR64 + VADDR32 + INST_OFFSET; the instruction
// offset is added to BOTH the global and the LDS address, so offset:0/16
// with the same registers copies 32 contiguous bytes per lane.
// LDS operand: wave-relative LDS byte offset = low 32 bits of the generic
// shared-memory pointer (LDS aperture keeps the offset in addr[31:0]).
// ---------------------------------------------------------------------------
__device__ inline void async_copy_32B(const _Float16* gbase, size_t elem_off,
                                      _Float16* lds_dst) {
  const unsigned lds  = (unsigned)(size_t)(void*)lds_dst;
  const unsigned goff = (unsigned)(elem_off * sizeof(_Float16));
  asm volatile("global_load_async_to_lds_b128 %0, %1, %2"
               :: "v"(lds), "v"(goff), "s"(gbase) : "memory");
  asm volatile("global_load_async_to_lds_b128 %0, %1, %2 offset:16"
               :: "v"(lds), "v"(goff), "s"(gbase) : "memory");
}

__device__ inline void wait_async_all() {
  asm volatile("s_wait_asynccnt 0x0" ::: "memory");
}

// ---------------------------------------------------------------------------
// Fragment packing (wave32, 16-bit operands, K=32): contiguous k = 16*(l>>4) + s in A and B
// ---------------------------------------------------------------------------

// A (16x32, MxK): lane m=l&15 is row m; half-wave (l>>4) selects K base 0 / 16.
__device__ inline v16h load_frag_a(const _Float16* p, int stride_h) {
  const int l  = threadIdx.x & 31;
  const int m  = l & 15;
  const int kh = (l >> 4) * 16;
  const unsigned* row = (const unsigned*)(p + m * stride_h + kh);
  union { v16h h; unsigned u[8]; } r;
#pragma unroll
  for (int i = 0; i < 8; ++i)
    r.u[i] = row[i];      // K pairs 0,2,...,14 (+kh)
  return r.h;
}

// B (32x16, KxN): lane n=l&15 is column n; half-wave selects K base 0 / 16.
// Storage must be K-contiguous per column: p + n*stride_h + k.
__device__ inline v16h load_frag_b(const _Float16* p, int stride_h) {
  const int l  = threadIdx.x & 31;
  const int n  = l & 15;
  const int kb = (l >> 4) * 16;
  const uint4* q4 = (const uint4*)(p + n * stride_h + kb);
  union { v16h h; uint4 u[2]; } r;
  r.u[0] = q4[0];
  r.u[1] = q4[1];
  return r.h;
}

// C/D (16x16 f32): lane holds column n=l&15; VGPR r is row r + (l>>4)*8.

// ---------------------------------------------------------------------------
// fp32 -> f16 convert (vectorized x4)
// ---------------------------------------------------------------------------
__global__ void cvt_f32_f16(const float4* __restrict__ src,
                            _Float16* __restrict__ dst, int n4) {
  int i = blockIdx.x * blockDim.x + threadIdx.x;
  if (i < n4) {
    float4 f = src[i];
    union { _Float16 h[4]; uint2 u; } p;
    p.h[0] = (_Float16)f.x; p.h[1] = (_Float16)f.y;
    p.h[2] = (_Float16)f.z; p.h[3] = (_Float16)f.w;
    ((uint2*)dst)[i] = p.u;
  }
}

// ---------------------------------------------------------------------------
// Tiled WMMA GEMM, double-buffered K loop with async A-panel staging.
// Workgroup: 128 threads (4 waves), 64x64 tile; wave = 32x32 (2x2 WMMA frags).
// MODE 0: scatter f16 q/k/v to [B,H,T,Dh] (q pre-scaled by 1/sqrt(Dh))
// MODE 1: f32 output to outf[M,N]
// ---------------------------------------------------------------------------
template <int MODE>
__global__ __launch_bounds__(128) void gemm_kernel(
    const _Float16* __restrict__ A, const _Float16* __restrict__ Bm,
    const float* __restrict__ bias, int N, int K,
    _Float16* __restrict__ qo, _Float16* __restrict__ ko,
    _Float16* __restrict__ vo, float* __restrict__ outf) {
  __shared__ alignas(16) _Float16 Atile[2][64 * 32];  // [row][k]
  __shared__ alignas(16) _Float16 Btile[2][64 * 32];  // [col][k] K-contig per col

  const int mblk = blockIdx.x;
  const int nblk = blockIdx.y;
  const int tid  = threadIdx.x;
  const int wave = tid >> 5;
  const int wm   = wave >> 1, wn = wave & 1;
  const int rowA = tid >> 1, segA = tid & 1;   // A staging: 32 B per thread
  const int kkB  = tid & 31, ncB  = tid >> 5;  // B staging: 16 halves per thread

  v8f acc00 = {}, acc01 = {}, acc10 = {}, acc11 = {};

  // ---- prologue: stage tile 0 into buffer 0 ----
  async_copy_32B(A, (size_t)(mblk * 64 + rowA) * K + segA * 16,
                 &Atile[0][rowA * 32 + segA * 16]);
  {
    const uint4* g = (const uint4*)(Bm + (size_t)kkB * N + nblk * 64 + ncB * 16);
    union { uint4 u[2]; _Float16 h[16]; } tb;
    tb.u[0] = g[0];
    tb.u[1] = g[1];
#pragma unroll
    for (int j = 0; j < 16; ++j)
      Btile[0][(ncB * 16 + j) * 32 + kkB] = tb.h[j];
  }
  wait_async_all();
  __syncthreads();

  for (int k0 = 0; k0 < K; k0 += 32) {
    const int cur = (k0 >> 5) & 1;
    const int nxt = cur ^ 1;
    const bool more = (k0 + 32) < K;

    uint4 w0 = {}, w1 = {};
    if (more) {
      // async DMA of next A panel overlaps the WMMAs below
      async_copy_32B(A, (size_t)(mblk * 64 + rowA) * K + (k0 + 32) + segA * 16,
                     &Atile[nxt][rowA * 32 + segA * 16]);
      // issue next B global loads now; scatter to LDS after compute
      const uint4* g =
          (const uint4*)(Bm + (size_t)(k0 + 32 + kkB) * N + nblk * 64 + ncB * 16);
      w0 = g[0];
      w1 = g[1];
    }

    v16h a0 = load_frag_a(&Atile[cur][(wm * 32 + 0)  * 32], 32);
    v16h a1 = load_frag_a(&Atile[cur][(wm * 32 + 16) * 32], 32);
    v16h b0 = load_frag_b(&Btile[cur][(wn * 32 + 0)  * 32], 32);
    v16h b1 = load_frag_b(&Btile[cur][(wn * 32 + 16) * 32], 32);
    acc00 = WMMA(a0, b0, acc00);
    acc01 = WMMA(a0, b1, acc01);
    acc10 = WMMA(a1, b0, acc10);
    acc11 = WMMA(a1, b1, acc11);

    if (more) {
      union { uint4 u[2]; _Float16 h[16]; } tb;
      tb.u[0] = w0;
      tb.u[1] = w1;
#pragma unroll
      for (int j = 0; j < 16; ++j)
        Btile[nxt][(ncB * 16 + j) * 32 + kkB] = tb.h[j];
    }
    wait_async_all();
    __syncthreads();
  }

  // ---- epilogue ----
  const int l     = tid & 31;
  const int cn    = l & 15;
  const int rbase = (l >> 4) * 8;
  const int row0  = mblk * 64 + wm * 32;
  const int col0  = nblk * 64 + wn * 32;

#pragma unroll
  for (int mt = 0; mt < 2; ++mt) {
#pragma unroll
    for (int nt = 0; nt < 2; ++nt) {
      v8f acc = (mt == 0) ? (nt == 0 ? acc00 : acc01) : (nt == 0 ? acc10 : acc11);
      const int col = col0 + nt * 16 + cn;
      const float bv = bias[col];
      if (MODE == 0) {
        const int which = col >> 10;           // 0:q 1:k 2:v
        const int c  = col & 1023;
        const int hh = c >> 6, dd = c & 63;
        _Float16* dst = (which == 0) ? qo : ((which == 1) ? ko : vo);
        const float scale = (which == 0) ? 0.125f : 1.0f;  // 1/sqrt(64) into q
#pragma unroll
        for (int r = 0; r < 8; ++r) {
          const int t  = row0 + mt * 16 + rbase + r;
          const int bb = t >> 11, tt = t & (kT - 1);
          dst[((size_t)((bb * kH + hh) * kT + tt) << 6) + dd] =
              (_Float16)((acc[r] + bv) * scale);
        }
      } else {
#pragma unroll
        for (int r = 0; r < 8; ++r) {
          const int t = row0 + mt * 16 + rbase + r;
          outf[(size_t)t * kD + col] = acc[r] + bv;
        }
      }
    }
  }
}

// ---------------------------------------------------------------------------
// Flash attention: grid (T/64, H, B), 128 threads (4 waves).
// Wave w owns q rows [qblk*64 + w*16, +16). KV streamed in 32-wide blocks.
// K tile staged via async global->LDS DMA; V staged transposed manually.
// ---------------------------------------------------------------------------
__global__ __launch_bounds__(128) void attn_kernel(
    const _Float16* __restrict__ q, const _Float16* __restrict__ k,
    const _Float16* __restrict__ v, _Float16* __restrict__ attn_out) {
  __shared__ alignas(16) _Float16 Ktile[32 * 64];   // [kv][d]
  __shared__ alignas(16) _Float16 VT[64 * 32];      // [d][kv]
  __shared__ alignas(16) _Float16 Ptile[4][16 * 32];

  const int tid  = threadIdx.x;
  const int wave = tid >> 5;
  const int l    = tid & 31;
  const int qblk = blockIdx.x;
  const int h    = blockIdx.y;
  const int b    = blockIdx.z;

  const size_t head_off = ((size_t)(b * kH + h)) * kT * kDh;
  const int qrow0 = qblk * 64 + wave * 16;

  // Q fragments (A layout) straight from global: rows contiguous by Dh=64
  const _Float16* qp = q + head_off + (size_t)qrow0 * kDh;
  const v16h aq0 = load_frag_a(qp, kDh);        // d 0..31
  const v16h aq1 = load_frag_a(qp + 32, kDh);   // d 32..63

  v8f o0 = {}, o1 = {}, o2 = {}, o3 = {};
  float mrow[8], lrow[8];
#pragma unroll
  for (int r = 0; r < 8; ++r) { mrow[r] = -1e30f; lrow[r] = 0.f; }

  const int cn    = l & 15;
  const int rbase = (l >> 4) * 8;
  const int rowS  = tid >> 2, segS = tid & 3;   // staging: 32 B per thread
  const int nblk  = qblk * 2 + 2;               // causal KV extent in 32-blocks

  for (int kb = 0; kb < nblk; ++kb) {
    const int kv0 = kb * 32;
    // K tile: straight copy -> async DMA to LDS
    async_copy_32B(k, head_off + (size_t)(kv0 + rowS) * kDh + segS * 16,
                   &Ktile[rowS * 64 + segS * 16]);
    // V tile: transpose scatter (VT[d][kv])
    {
      const uint4* gv =
          (const uint4*)(v + head_off + (size_t)(kv0 + rowS) * kDh + segS * 16);
      union { uint4 u[2]; _Float16 h[16]; } tv;
      tv.u[0] = gv[0];
      tv.u[1] = gv[1];
#pragma unroll
      for (int j = 0; j < 16; ++j)
        VT[(segS * 16 + j) * 32 + rowS] = tv.h[j];
    }
    wait_async_all();
    __syncthreads();

    // S = Q * K^T  (two 16-col tiles, d split 0..31 / 32..63)
    v8f s0 = {}, s1 = {};
    {
      v16h b00 = load_frag_b(Ktile, 64);
      v16h b01 = load_frag_b(Ktile + 32, 64);
      v16h b10 = load_frag_b(Ktile + 16 * 64, 64);
      v16h b11 = load_frag_b(Ktile + 16 * 64 + 32, 64);
      s0 = WMMA(aq0, b00, s0);
      s0 = WMMA(aq1, b01, s0);
      s1 = WMMA(aq0, b10, s1);
      s1 = WMMA(aq1, b11, s1);
    }

    // online softmax over the 32 new columns; P -> per-wave LDS tile (f16)
    _Float16* P = &Ptile[wave][0];
#pragma unroll
    for (int r = 0; r < 8; ++r) {
      const int qg = qrow0 + rbase + r;
      const float x0 = ((kv0 + cn)      <= qg) ? s0[r] : -1e30f;
      const float x1 = ((kv0 + 16 + cn) <= qg) ? s1[r] : -1e30f;
      float mx = fmaxf(x0, x1);
#pragma unroll
      for (int off = 8; off >= 1; off >>= 1)
        mx = fmaxf(mx, __shfl_xor(mx, off, 32));
      mx = fmaxf(mx, mrow[r]);
      const float p0 = __expf(x0 - mx);
      const float p1 = __expf(x1 - mx);
      const float corr = __expf(mrow[r] - mx);
      mrow[r] = mx;
      float sum = p0 + p1;
#pragma unroll
      for (int off = 8; off >= 1; off >>= 1)
        sum += __shfl_xor(sum, off, 32);
      lrow[r] = lrow[r] * corr + sum;
      o0[r] *= corr; o1[r] *= corr; o2[r] *= corr; o3[r] *= corr;
      const int m = rbase + r;
      P[m * 32 + cn]      = (_Float16)p0;
      P[m * 32 + 16 + cn] = (_Float16)p1;
    }

    // O += P * V   (LDS DS ops are in-order within a wave)
    {
      v16h ap  = load_frag_a(P, 32);
      v16h bv0 = load_frag_b(VT + 0 * 16 * 32, 32);
      v16h bv1 = load_frag_b(VT + 1 * 16 * 32, 32);
      v16h bv2 = load_frag_b(VT + 2 * 16 * 32, 32);
      v16h bv3 = load_frag_b(VT + 3 * 16 * 32, 32);
      o0 = WMMA(ap, bv0, o0);
      o1 = WMMA(ap, bv1, o1);
      o2 = WMMA(ap, bv2, o2);
      o3 = WMMA(ap, bv3, o3);
    }
    __syncthreads();
  }

  // normalize and store f16 attention output in [B*T, D_MODEL] layout
  const int qcol0 = h * kDh;
#pragma unroll
  for (int r = 0; r < 8; ++r) {
    const float inv = 1.0f / lrow[r];
    const int t = qrow0 + rbase + r;
    const size_t rowoff = ((size_t)(b * kT + t)) * kD;
    attn_out[rowoff + qcol0 + 0  + cn] = (_Float16)(o0[r] * inv);
    attn_out[rowoff + qcol0 + 16 + cn] = (_Float16)(o1[r] * inv);
    attn_out[rowoff + qcol0 + 32 + cn] = (_Float16)(o2[r] * inv);
    attn_out[rowoff + qcol0 + 48 + cn] = (_Float16)(o3[r] * inv);
  }
}

// ---------------------------------------------------------------------------
// Launch
// ---------------------------------------------------------------------------
extern "C" void kernel_launch(void* const* d_in, const int* in_sizes, int n_in,
                              void* d_out, int out_size, void* d_ws, size_t ws_size,
                              hipStream_t stream) {
  const float* x      = (const float*)d_in[0];
  const float* W_qkv  = (const float*)d_in[1];
  const float* b_qkv  = (const float*)d_in[2];
  const float* W_proj = (const float*)d_in[3];
  const float* b_proj = (const float*)d_in[4];
  float* out = (float*)d_out;

  // workspace layout (f16), total 24M halves = 48 MB
  _Float16* xh     = (_Float16*)d_ws;                       // 4096*1024
  _Float16* wqkvh  = xh     + (size_t)kM * kD;              // 1024*3072
  _Float16* wprojh = wqkvh  + (size_t)kD * k3D;             // 1024*1024
  _Float16* qh     = wprojh + (size_t)kD * kD;              // [B,H,T,Dh]
  _Float16* kh     = qh     + (size_t)kM * kD;
  _Float16* vh     = kh     + (size_t)kM * kD;
  _Float16* ah     = vh     + (size_t)kM * kD;              // attn out [B*T, D]

  const int nx = kM * kD, nwq = kD * k3D, nwp = kD * kD;
  cvt_f32_f16<<<dim3((nx / 4 + 255) / 256), 256, 0, stream>>>((const float4*)x, xh, nx / 4);
  cvt_f32_f16<<<dim3((nwq / 4 + 255) / 256), 256, 0, stream>>>((const float4*)W_qkv, wqkvh, nwq / 4);
  cvt_f32_f16<<<dim3((nwp / 4 + 255) / 256), 256, 0, stream>>>((const float4*)W_proj, wprojh, nwp / 4);

  // qkv = x @ W_qkv + b_qkv  -> scatter to q/k/v head layout
  gemm_kernel<0><<<dim3(kM / 64, k3D / 64), 128, 0, stream>>>(
      xh, wqkvh, b_qkv, k3D, kD, qh, kh, vh, nullptr);

  // flash attention
  attn_kernel<<<dim3(kT / 64, kH, kB), 128, 0, stream>>>(qh, kh, vh, ah);

  // out = attn @ W_proj + b_proj  (f32)
  gemm_kernel<1><<<dim3(kM / 64, kD / 64), 128, 0, stream>>>(
      ah, wprojh, b_proj, kD, kD, nullptr, nullptr, nullptr, out);

  (void)in_sizes; (void)n_in; (void)out_size; (void)ws_size;
}
